// HyperbolicLMHead_2473901163016
// MI455X (gfx1250) — matrix operand order repaired and checked by reading, not compiled, not run on hardware
//
#include <hip/hip_runtime.h>
#include <hip/hip_bf16.h>
#include <math.h>

// ---------------- problem constants ----------------
#define M_DIM 4096        // 2 * 2048 rows of x
#define D_DIM 1024        // hidden dim (K)
#define V_DIM 50257       // vocab (N)
#define HC    1.0f
#define HEPS  1e-6f
#define MAXNORM 0.999999f // (1 - EPS)/sqrt(C)

typedef __attribute__((ext_vector_type(16))) __bf16 v16bf;
typedef __attribute__((ext_vector_type(8)))  float  v8f;

// round-to-nearest-even f32 -> bf16 (as ushort)
static __device__ __forceinline__ unsigned short f2bf(float f) {
  unsigned int u = __float_as_uint(f);
  u += 0x7FFFu + ((u >> 16) & 1u);
  return (unsigned short)(u >> 16);
}
static __device__ __forceinline__ float bf2f(unsigned short h) {
  return __uint_as_float(((unsigned int)h) << 16);
}

// ---------------------------------------------------------------------------
// Phase 1: per-row Poincare-ball projection + bf16 conversion.
// One 256-thread block per row of 1024 floats (each thread owns one float4).
// Writes bf16 row and the squared norm of the *converted* row (keeps the
// x2 + w2 - 2*cross identity consistent with the bf16 GEMM).
// ---------------------------------------------------------------------------
__global__ __launch_bounds__(256) void project_rows_kernel(
    const float* __restrict__ in, unsigned short* __restrict__ out_bf,
    float* __restrict__ out_sq) {
  const int r = blockIdx.x;
  const int t = threadIdx.x;
  const float4 v = ((const float4*)(in + (size_t)r * D_DIM))[t];

  __shared__ float red[256];
  float ss = v.x * v.x + v.y * v.y + v.z * v.z + v.w * v.w;
  red[t] = ss;
  __syncthreads();
  for (int s = 128; s > 0; s >>= 1) {
    if (t < s) red[t] += red[t + s];
    __syncthreads();
  }
  const float norm  = fmaxf(sqrtf(red[0]), HEPS);
  const float scale = fminf(MAXNORM / norm, 1.0f);

  unsigned short b0 = f2bf(v.x * scale);
  unsigned short b1 = f2bf(v.y * scale);
  unsigned short b2 = f2bf(v.z * scale);
  unsigned short b3 = f2bf(v.w * scale);

  // squared norm of the bf16-rounded row
  float c0 = bf2f(b0), c1 = bf2f(b1), c2 = bf2f(b2), c3 = bf2f(b3);
  float ss2 = c0 * c0 + c1 * c1 + c2 * c2 + c3 * c3;
  __syncthreads();
  red[t] = ss2;
  __syncthreads();
  for (int s = 128; s > 0; s >>= 1) {
    if (t < s) red[t] += red[t + s];
    __syncthreads();
  }
  if (t == 0) out_sq[r] = red[0];

  ushort2 lo = make_ushort2(b0, b1);
  ushort2 hi = make_ushort2(b2, b3);
  ushort2* dst = (ushort2*)(out_bf + (size_t)r * D_DIM + t * 4);
  dst[0] = lo;
  dst[1] = hi;
}

// ---------------------------------------------------------------------------
// Phase 2: bf16 WMMA GEMM + hyperbolic distance epilogue.
// Block = 256 threads = 8 waves; block tile 128(M) x 64(N).
// Wave tile 32x32 -> 4 f32 accumulators, K loop of 32 with one
// v_wmma_f32_16x16x32_bf16 per (mi,ni) pair.
// ---------------------------------------------------------------------------
union FragU {
  uint4 u[2];
  v16bf v;
};

static __device__ __forceinline__ void epilogue_store(
    const v8f& acc, const float* x2r, float w2n, int m0, int n,
    float* __restrict__ out) {
  if (n >= V_DIM) return;
#pragma unroll
  for (int i = 0; i < 8; ++i) {
    float cross = acc[i];
    float x2m   = x2r[i];
    float sq    = fmaxf(x2m + w2n - 2.0f * cross, 0.0f);
    float den   = fmaxf((1.0f - HC * x2m) * (1.0f - HC * w2n), HEPS);
    float arg   = fmaxf(1.0f + 2.0f * HC * sq / den, 1.0f + HEPS);
    float d     = acoshf(arg);
    __builtin_nontemporal_store(-d * d, out + (size_t)(m0 + i) * V_DIM + n);
  }
}

__global__ __launch_bounds__(256) void hyper_lmhead_gemm_kernel(
    const unsigned short* __restrict__ xb,  // [M_DIM][D_DIM] bf16
    const unsigned short* __restrict__ wb,  // [V_DIM][D_DIM] bf16
    const float* __restrict__ x2,           // [M_DIM]
    const float* __restrict__ w2,           // [V_DIM]
    float* __restrict__ out) {              // [M_DIM][V_DIM]
  const int lane = threadIdx.x & 31;
  const int wave = threadIdx.x >> 5;
  const int wrow = wave & 3;   // 4 waves along M
  const int wcol = wave >> 2;  // 2 waves along N

  const int m_wave = blockIdx.y * 128 + wrow * 32;
  const int n_wave = blockIdx.x * 64 + wcol * 32;

  const int hrow = lane & 15;   // row (A) / col (B) within 16
  const int half = lane >> 4;   // K-half selector

  // A rows for the two 16-row sub-tiles
  const unsigned short* pa0 =
      xb + (size_t)(m_wave + hrow) * D_DIM + half * 16;
  const unsigned short* pa1 =
      xb + (size_t)(m_wave + 16 + hrow) * D_DIM + half * 16;

  // B columns (vocab rows), clamped for the ragged N edge
  const int n0 = n_wave + hrow;
  const int n1 = n_wave + 16 + hrow;
  const int n0c = (n0 < V_DIM) ? n0 : (V_DIM - 1);
  const int n1c = (n1 < V_DIM) ? n1 : (V_DIM - 1);
  const unsigned short* pb0 = wb + (size_t)n0c * D_DIM + half * 16;
  const unsigned short* pb1 = wb + (size_t)n1c * D_DIM + half * 16;

  v8f acc00 = {}, acc01 = {}, acc10 = {}, acc11 = {};

  for (int k0 = 0; k0 < D_DIM; k0 += 32) {
    // prefetch next K-step of the B stream (global_prefetch_b8)
    if (k0 + 32 < D_DIM) {
      __builtin_prefetch(pb0 + k0 + 32, 0, 1);
      __builtin_prefetch(pb1 + k0 + 32, 0, 1);
    }

    FragU a0, a1, b0, b1;
    // A 16x32 in this code's contiguous k order (same as B): lanes 0-15 hold K 0..15,
    // lanes 16-31 hold K 16..31 (half*16 baked into pa).
    a0.u[0] = *(const uint4*)(pa0 + k0);
    a0.u[1] = *(const uint4*)(pa0 + k0 + 8);
    a1.u[0] = *(const uint4*)(pa1 + k0);
    a1.u[1] = *(const uint4*)(pa1 + k0 + 8);
    // B 32x16 in the same contiguous k order: lane = N, lanes 0-15 hold K 0..15,
    // lanes 16-31 hold K 16..31 (half*16 baked into pb).
    b0.u[0] = *(const uint4*)(pb0 + k0);
    b0.u[1] = *(const uint4*)(pb0 + k0 + 8);
    b1.u[0] = *(const uint4*)(pb1 + k0);
    b1.u[1] = *(const uint4*)(pb1 + k0 + 8);

    acc00 = __builtin_amdgcn_wmma_f32_16x16x32_bf16(
        false, a0.v, false, b0.v, (short)0, acc00, false, false);
    acc01 = __builtin_amdgcn_wmma_f32_16x16x32_bf16(
        false, a0.v, false, b1.v, (short)0, acc01, false, false);
    acc10 = __builtin_amdgcn_wmma_f32_16x16x32_bf16(
        false, a1.v, false, b0.v, (short)0, acc10, false, false);
    acc11 = __builtin_amdgcn_wmma_f32_16x16x32_bf16(
        false, a1.v, false, b1.v, (short)0, acc11, false, false);
  }

  // C/D layout: VGPR i -> M = i + (lane>=16 ? 8 : 0), N = lane & 15.
  const int rbase0 = m_wave + half * 8;
  const int rbase1 = rbase0 + 16;
  float x2r0[8], x2r1[8];
#pragma unroll
  for (int i = 0; i < 8; ++i) {
    x2r0[i] = x2[rbase0 + i];
    x2r1[i] = x2[rbase1 + i];
  }
  const float w2n0 = (n0 < V_DIM) ? w2[n0] : 0.0f;
  const float w2n1 = (n1 < V_DIM) ? w2[n1] : 0.0f;

  epilogue_store(acc00, x2r0, w2n0, rbase0, n0, out);
  epilogue_store(acc01, x2r0, w2n1, rbase0, n1, out);
  epilogue_store(acc10, x2r1, w2n0, rbase1, n0, out);
  epilogue_store(acc11, x2r1, w2n1, rbase1, n1, out);
}

// ---------------------------------------------------------------------------
// Launch
// ---------------------------------------------------------------------------
extern "C" void kernel_launch(void* const* d_in, const int* in_sizes, int n_in,
                              void* d_out, int out_size, void* d_ws,
                              size_t ws_size, hipStream_t stream) {
  (void)in_sizes; (void)n_in; (void)out_size; (void)ws_size;
  const float* x = (const float*)d_in[0];   // [2,2048,1024]
  const float* w = (const float*)d_in[1];   // [50257,1024]
  float* out = (float*)d_out;               // [2,2048,50257]

  // workspace layout
  char* ws = (char*)d_ws;
  const size_t XB_BYTES = (size_t)M_DIM * D_DIM * sizeof(unsigned short); // 8 MiB
  const size_t WB_BYTES = (size_t)V_DIM * D_DIM * sizeof(unsigned short); // ~98 MiB
  unsigned short* xb = (unsigned short*)ws;
  unsigned short* wb = (unsigned short*)(ws + XB_BYTES);
  float* x2 = (float*)(ws + XB_BYTES + WB_BYTES);
  float* w2 = x2 + M_DIM;

  project_rows_kernel<<<M_DIM, 256, 0, stream>>>(x, xb, x2);
  project_rows_kernel<<<V_DIM, 256, 0, stream>>>(w, wb, w2);

  dim3 grid((V_DIM + 63) / 64, M_DIM / 128);  // (786, 32)
  hyper_lmhead_gemm_kernel<<<grid, 256, 0, stream>>>(xb, wb, x2, w2, out);
}
